// MoEFFN_53901839565260
// MI455X (gfx1250) — matrix operand order repaired and checked by reading, not compiled, not run on hardware
//
#include <hip/hip_runtime.h>

// ---------------- problem constants ----------------
#define T_TOK 8192
#define DM    1024
#define DFF   4096
#define NE    8
#define TOPK  2

// ---------------- GEMM tiling ----------------
#define BM 64
#define BN 256
#define BK 32
#define ASTRIDE (BK + 8)          // 40 bf16 = 80 B rows -> 16B-aligned segments
#define RBLOCKS (T_TOK / 8)

// CDNA5 async global->LDS path (ASYNCcnt). Guarded: falls back to sync copy.
#if __has_builtin(__builtin_amdgcn_global_load_async_to_lds_b128) && \
    __has_builtin(__builtin_amdgcn_s_wait_asynccnt)
#define HAS_ASYNC_LDS 1
#endif

typedef __attribute__((ext_vector_type(16))) __bf16 v16bf;
typedef __attribute__((ext_vector_type(8)))  float  v8f;
typedef __attribute__((ext_vector_type(4)))  int    v4i;

union FragBF { v16bf v; uint4 q[2]; };

__device__ __forceinline__ unsigned short f2bf(float f) {
  unsigned u = __float_as_uint(f);
  u += 0x7FFFu + ((u >> 16) & 1u);          // round-to-nearest-even
  return (unsigned short)(u >> 16);
}

__device__ __forceinline__ float gelu_exact(float x) {
  return 0.5f * x * (1.0f + erff(x * 0.70710678118654752f));
}

// one 16B global->LDS move; async (ASYNCcnt-tracked) when available
__device__ __forceinline__ void copy_b128(unsigned short* dstLds, const unsigned short* srcG) {
#ifdef HAS_ASYNC_LDS
  __builtin_amdgcn_global_load_async_to_lds_b128(
      (__attribute__((address_space(1))) v4i*)(unsigned long long)srcG,
      (__attribute__((address_space(3))) v4i*)(unsigned)(unsigned long long)dstLds,
      0, 0);
#else
  *(uint4*)dstLds = *(const uint4*)srcG;
#endif
}
// 5 b128 asyncs issued per thread per stage -> waiting <=5 releases the older stage
__device__ __forceinline__ void wait_async_prev() {
#ifdef HAS_ASYNC_LDS
  __builtin_amdgcn_s_wait_asynccnt(5);
#endif
}
__device__ __forceinline__ void wait_async_all() {
#ifdef HAS_ASYNC_LDS
  __builtin_amdgcn_s_wait_asynccnt(0);
#endif
}

// A-matrix 16x32 bf16 fragment: lane L -> M=L%16; elem i -> K=((i>=8)?16:0)+8*laneHi+(i&7)
__device__ __forceinline__ v16bf load_frag_a(const unsigned short* As, int mrow, int laneHi) {
  FragBF f;
  const unsigned short* base = As + mrow * ASTRIDE + laneHi * 8;
  f.q[0] = *(const uint4*)(base);
  f.q[1] = *(const uint4*)(base + 16);
  return f.v;
}
// B-matrix 32x16 bf16 fragment: lane L -> N=L%16; elem i -> K=((i>=8)?16:0)+8*laneHi+(i&7)
__device__ __forceinline__ v16bf load_frag_b(const unsigned short* Bs, int ncol, int laneHi) {
  FragBF f;
  const unsigned short* base = Bs + ncol * ASTRIDE + laneHi * 8;
  f.q[0] = *(const uint4*)(base);
  f.q[1] = *(const uint4*)(base + 16);
  return f.v;
}

// ---------------- setup kernels ----------------
__global__ void zero_init_kernel(float* __restrict__ out, int n, int* __restrict__ counts,
                                 float* __restrict__ zpage) {
  int i = blockIdx.x * blockDim.x + threadIdx.x;
  if (i < n) out[i] = 0.0f;
  if (i < NE) counts[i] = 0;
  if (i < 64) zpage[i] = 0.0f;
}

// generic fp32 -> bf16 cast, 4 elems/thread (used for x once, and each expert's w1/w2 once)
__global__ void cast_bf16_kernel(const float* __restrict__ src, unsigned short* __restrict__ dst) {
  size_t i = ((size_t)blockIdx.x * blockDim.x + threadIdx.x) * 4;
  float4 f = *(const float4*)(src + i);
  ushort4 o;
  o.x = f2bf(f.x); o.y = f2bf(f.y); o.z = f2bf(f.z); o.w = f2bf(f.w);
  *(ushort4*)(dst + i) = o;
}

// ---------------- router: logits, softmax, top-2, gather lists, aux partials ----------------
__global__ void router_kernel(const float* __restrict__ x, const float* __restrict__ gate,
                              int* __restrict__ counts, int* __restrict__ idxList,
                              float* __restrict__ wgtList, float* __restrict__ partial) {
  __shared__ float gs[NE * DM];       // 32 KB: full gate matrix
  __shared__ float probsS[8][NE];
  const int tid = threadIdx.x;
  for (int i = tid; i < NE * DM; i += 256) gs[i] = gate[i];
  __syncthreads();

  const int wv = tid >> 5, lane = tid & 31;
  const int t = blockIdx.x * 8 + wv;           // one token per wave32
  const float* xr = x + (size_t)t * DM;

  float acc[NE];
#pragma unroll
  for (int e = 0; e < NE; ++e) acc[e] = 0.0f;
  for (int d = lane; d < DM; d += 32) {
    float xv = xr[d];
#pragma unroll
    for (int e = 0; e < NE; ++e) acc[e] += xv * gs[e * DM + d];
  }
#pragma unroll
  for (int m = 16; m >= 1; m >>= 1) {
#pragma unroll
    for (int e = 0; e < NE; ++e) acc[e] += __shfl_xor(acc[e], m, 32);
  }
  float mx = acc[0];
#pragma unroll
  for (int e = 1; e < NE; ++e) mx = fmaxf(mx, acc[e]);
  float p[NE], se = 0.0f;
#pragma unroll
  for (int e = 0; e < NE; ++e) { p[e] = expf(acc[e] - mx); se += p[e]; }
  float inv = 1.0f / se;
#pragma unroll
  for (int e = 0; e < NE; ++e) p[e] *= inv;

  int i1 = 0;
#pragma unroll
  for (int e = 1; e < NE; ++e) if (p[e] > p[i1]) i1 = e;
  int i2 = (i1 == 0) ? 1 : 0;
#pragma unroll
  for (int e = 0; e < NE; ++e) if (e != i1 && p[e] > p[i2]) i2 = e;
  float s12 = p[i1] + p[i2];

  if (lane == 0) {
    int pos1 = atomicAdd(&counts[i1], 1);
    idxList[i1 * T_TOK + pos1] = t;
    wgtList[i1 * T_TOK + pos1] = p[i1] / s12;
    int pos2 = atomicAdd(&counts[i2], 1);
    idxList[i2 * T_TOK + pos2] = t;
    wgtList[i2 * T_TOK + pos2] = p[i2] / s12;
#pragma unroll
    for (int e = 0; e < NE; ++e) probsS[wv][e] = p[e];
  }
  __syncthreads();
  if (tid < NE) {   // fixed-order per-block reduction -> deterministic aux loss
    float sum = 0.0f;
#pragma unroll
    for (int k = 0; k < 8; ++k) sum += probsS[k][tid];
    partial[blockIdx.x * NE + tid] = sum;
  }
}

__global__ void aux_kernel(const int* __restrict__ counts, const float* __restrict__ partial,
                           float* __restrict__ auxOut) {
  __shared__ float ps[NE];
  const int tid = threadIdx.x;
  if (tid < NE) {
    float s = 0.0f;
    for (int b = 0; b < RBLOCKS; ++b) s += partial[b * NE + tid];
    ps[tid] = s;
  }
  __syncthreads();
  if (tid == 0) {
    float aux = 0.0f;
    for (int e = 0; e < NE; ++e) {
      float frac = (float)counts[e] / (float)(T_TOK * TOPK);
      float avgp = ps[e] / (float)T_TOK;
      aux += frac * avgp;
    }
    *auxOut = aux * (float)NE;
  }
}

// ---------------- pass 1: H = gelu(gather(X) @ W1e^T + b1e), bf16 ----------------
__global__ void __launch_bounds__(256, 2)
expert_ffn1_kernel(const unsigned short* __restrict__ xbf,
                   const unsigned short* __restrict__ w1bf, const float* __restrict__ b1e,
                   const int* __restrict__ idxList, const int* __restrict__ countPtr,
                   const unsigned short* __restrict__ zp,
                   unsigned short* __restrict__ H) {
  const int cnt = *countPtr;                 // uniform scalar load
  const int rowBase = blockIdx.y * BM;
  if (rowBase >= cnt) return;                // block-uniform: EXEC all-1s at every WMMA
  const int colBase = blockIdx.x * BN;

  __shared__ unsigned short As[2][BM * ASTRIDE];   // 2 x  5.0 KB
  __shared__ unsigned short Bs[2][BN * ASTRIDE];   // 2 x 20.0 KB

  const int tid = threadIdx.x;
  const int lane = tid & 31, wv = tid >> 5;
  const int laneHi = (lane >> 4) & 1, lm = lane & 15;

  v8f acc[4][2];
#pragma unroll
  for (int ms = 0; ms < 4; ++ms)
#pragma unroll
    for (int c = 0; c < 2; ++c)
#pragma unroll
      for (int j = 0; j < 8; ++j) acc[ms][c][j] = 0.0f;

  // A staging: 64x32 tile, one b128 per thread per stage
  const int ar = tid >> 2;
  const int aseg = (tid & 3) * 8;
  const int agr = rowBase + ar;
  const bool aval = (agr < cnt);
  const int atok = aval ? idxList[agr] : 0;
  const unsigned short* agp = aval ? (xbf + (size_t)atok * DM + aseg) : zp;
  const int astep = aval ? BK : 0;           // OOB rows keep reading the zero page
  // B staging: 256x32 tile, four b128 per thread per stage
  const unsigned short* bgp = w1bf + (size_t)(colBase + tid) * DM;

  // prologue: stage tile 0 into buffer 0
  copy_b128(&As[0][ar * ASTRIDE + aseg], agp);
#pragma unroll
  for (int s = 0; s < 4; ++s) copy_b128(&Bs[0][tid * ASTRIDE + s * 8], bgp + s * 8);
  const unsigned short* agpn = agp + astep;

  for (int kt = 0; kt < DM; kt += BK) {
    const int buf = (kt / BK) & 1;
    if (kt + BK < DM) {                      // stage next tile into the other buffer
      copy_b128(&As[buf ^ 1][ar * ASTRIDE + aseg], agpn);
#pragma unroll
      for (int s = 0; s < 4; ++s)
        copy_b128(&Bs[buf ^ 1][tid * ASTRIDE + s * 8], bgp + kt + BK + s * 8);
      agpn += astep;
      wait_async_prev();                     // release current tile's 5 asyncs
    } else {
      wait_async_all();
    }
    __syncthreads();

    const unsigned short* Ab = As[buf];
    const unsigned short* Bb = Bs[buf];
    v16bf b0 = load_frag_b(Bb, wv * 32 + lm, laneHi);
    v16bf b1 = load_frag_b(Bb, wv * 32 + 16 + lm, laneHi);
#pragma unroll
    for (int ms = 0; ms < 4; ++ms) {
      v16bf a = load_frag_a(Ab, ms * 16 + lm, laneHi);
      acc[ms][0] = __builtin_amdgcn_wmma_f32_16x16x32_bf16(
          false, a, false, b0, (short)0, acc[ms][0], false, false);
      acc[ms][1] = __builtin_amdgcn_wmma_f32_16x16x32_bf16(
          false, a, false, b1, (short)0, acc[ms][1], false, false);
    }
    __syncthreads();
  }

#pragma unroll
  for (int c = 0; c < 2; ++c) {
    const int n = colBase + wv * 32 + c * 16 + lm;
    const float bias = b1e[n];
#pragma unroll
    for (int ms = 0; ms < 4; ++ms) {
#pragma unroll
      for (int j = 0; j < 8; ++j) {
        int r = rowBase + ms * 16 + laneHi * 8 + j;  // C/D layout: M = j + 8*laneHi
        if (r < cnt) {
          float h = gelu_exact(acc[ms][c][j] + bias);
          H[(size_t)r * DFF + n] = f2bf(h);
        }
      }
    }
  }
}

// ---------------- pass 2: out[tok] += wgt * (H @ W2e^T + b2e) ----------------
__global__ void __launch_bounds__(256, 2)
expert_ffn2_kernel(const unsigned short* __restrict__ H,
                   const unsigned short* __restrict__ w2bf, const float* __restrict__ b2e,
                   const int* __restrict__ idxList, const float* __restrict__ wgtList,
                   const int* __restrict__ countPtr,
                   const unsigned short* __restrict__ zp,
                   float* __restrict__ out) {
  const int cnt = *countPtr;
  const int rowBase = blockIdx.y * BM;
  if (rowBase >= cnt) return;
  const int colBase = blockIdx.x * BN;

  __shared__ unsigned short As[2][BM * ASTRIDE];
  __shared__ unsigned short Bs[2][BN * ASTRIDE];

  const int tid = threadIdx.x;
  const int lane = tid & 31, wv = tid >> 5;
  const int laneHi = (lane >> 4) & 1, lm = lane & 15;

  v8f acc[4][2];
#pragma unroll
  for (int ms = 0; ms < 4; ++ms)
#pragma unroll
    for (int c = 0; c < 2; ++c)
#pragma unroll
      for (int j = 0; j < 8; ++j) acc[ms][c][j] = 0.0f;

  const int ar = tid >> 2;
  const int aseg = (tid & 3) * 8;
  const int agr = rowBase + ar;
  const bool aval = (agr < cnt);
  const unsigned short* agp = aval ? (H + (size_t)agr * DFF + aseg) : zp;
  const int astep = aval ? BK : 0;
  const unsigned short* bgp = w2bf + (size_t)(colBase + tid) * DFF;

  copy_b128(&As[0][ar * ASTRIDE + aseg], agp);
#pragma unroll
  for (int s = 0; s < 4; ++s) copy_b128(&Bs[0][tid * ASTRIDE + s * 8], bgp + s * 8);
  const unsigned short* agpn = agp + astep;

  for (int kt = 0; kt < DFF; kt += BK) {
    const int buf = (kt / BK) & 1;
    if (kt + BK < DFF) {
      copy_b128(&As[buf ^ 1][ar * ASTRIDE + aseg], agpn);
#pragma unroll
      for (int s = 0; s < 4; ++s)
        copy_b128(&Bs[buf ^ 1][tid * ASTRIDE + s * 8], bgp + kt + BK + s * 8);
      agpn += astep;
      wait_async_prev();
    } else {
      wait_async_all();
    }
    __syncthreads();

    const unsigned short* Ab = As[buf];
    const unsigned short* Bb = Bs[buf];
    v16bf b0 = load_frag_b(Bb, wv * 32 + lm, laneHi);
    v16bf b1 = load_frag_b(Bb, wv * 32 + 16 + lm, laneHi);
#pragma unroll
    for (int ms = 0; ms < 4; ++ms) {
      v16bf a = load_frag_a(Ab, ms * 16 + lm, laneHi);
      acc[ms][0] = __builtin_amdgcn_wmma_f32_16x16x32_bf16(
          false, a, false, b0, (short)0, acc[ms][0], false, false);
      acc[ms][1] = __builtin_amdgcn_wmma_f32_16x16x32_bf16(
          false, a, false, b1, (short)0, acc[ms][1], false, false);
    }
    __syncthreads();
  }

#pragma unroll
  for (int c = 0; c < 2; ++c) {
    const int n = colBase + wv * 32 + c * 16 + lm;
    const float bias = b2e[n];
#pragma unroll
    for (int ms = 0; ms < 4; ++ms) {
#pragma unroll
      for (int j = 0; j < 8; ++j) {
        int r = rowBase + ms * 16 + laneHi * 8 + j;
        if (r < cnt) {
          int t = idxList[r];
          float y = (acc[ms][c][j] + bias) * wgtList[r];
          float* op = out + (size_t)t * DM + n;  // race-free: (t,n) unique in-kernel,
          *op += y;                              // experts serialized on the stream
        }
      }
    }
  }
}

// ---------------- host launch ----------------
extern "C" void kernel_launch(void* const* d_in, const int* in_sizes, int n_in,
                              void* d_out, int out_size, void* d_ws, size_t ws_size,
                              hipStream_t stream) {
  (void)in_sizes; (void)n_in; (void)ws_size;
  const float* x    = (const float*)d_in[0];
  const float* gate = (const float*)d_in[1];
  const float* w1   = (const float*)d_in[2];
  const float* b1   = (const float*)d_in[3];
  const float* w2   = (const float*)d_in[4];
  const float* b2   = (const float*)d_in[5];
  float* out = (float*)d_out;

  char* ws = (char*)d_ws;
  size_t off = 0;
  unsigned short* xbf  = (unsigned short*)(ws + off); off += (size_t)T_TOK * DM  * 2;  // 16 MB
  unsigned short* H    = (unsigned short*)(ws + off); off += (size_t)T_TOK * DFF * 2;  // 64 MB
  unsigned short* w1bf = (unsigned short*)(ws + off); off += (size_t)DFF * DM * 2;     //  8 MB
  unsigned short* w2bf = (unsigned short*)(ws + off); off += (size_t)DM * DFF * 2;     //  8 MB
  int*   counts  = (int*)(ws + off);   off += 256;
  int*   idxList = (int*)(ws + off);   off += (size_t)NE * T_TOK * 4;
  float* wgtList = (float*)(ws + off); off += (size_t)NE * T_TOK * 4;
  float* partial = (float*)(ws + off); off += (size_t)RBLOCKS * NE * 4;
  float* zpage   = (float*)(ws + off); off += 256;

  zero_init_kernel<<<(out_size + 255) / 256, 256, 0, stream>>>(out, out_size, counts, zpage);
  cast_bf16_kernel<<<(T_TOK * DM) / 1024, 256, 0, stream>>>(x, xbf);
  router_kernel<<<RBLOCKS, 256, 0, stream>>>(x, gate, counts, idxList, wgtList, partial);
  aux_kernel<<<1, 32, 0, stream>>>(counts, partial, out + (size_t)T_TOK * DM);

  for (int e = 0; e < NE; ++e) {
    cast_bf16_kernel<<<(DFF * DM) / 1024, 256, 0, stream>>>(w1 + (size_t)e * DFF * DM, w1bf);
    cast_bf16_kernel<<<(DM * DFF) / 1024, 256, 0, stream>>>(w2 + (size_t)e * DM * DFF, w2bf);
    expert_ffn1_kernel<<<dim3(DFF / BN, T_TOK / BM), 256, 0, stream>>>(
        xbf, w1bf, b1 + (size_t)e * DFF,
        idxList + (size_t)e * T_TOK, counts + e, (const unsigned short*)zpage, H);
    expert_ffn2_kernel<<<dim3(DM / BN, T_TOK / BM), 256, 0, stream>>>(
        H, w2bf, b2 + (size_t)e * DM,
        idxList + (size_t)e * T_TOK, wgtList + (size_t)e * T_TOK, counts + e,
        (const unsigned short*)zpage, out);
  }
}
